// AttentionBlock_57346403336339
// MI455X (gfx1250) — matrix operand order repaired and checked by reading, not compiled, not run on hardware
//
#include <hip/hip_runtime.h>

typedef __attribute__((ext_vector_type(16))) _Float16 v16h;
typedef __attribute__((ext_vector_type(8)))  _Float16 v8h;
typedef __attribute__((ext_vector_type(4)))  _Float16 v4h;
typedef __attribute__((ext_vector_type(2)))  _Float16 v2h;
typedef __attribute__((ext_vector_type(8)))  float    v8f;
typedef __attribute__((ext_vector_type(4)))  float    f4;

#define DEV __device__ __forceinline__

// ---------------- wave32 reductions ----------------
DEV float wsum32(float v){
  #pragma unroll
  for (int m = 1; m < 32; m <<= 1) v += __shfl_xor(v, m, 32);
  return v;
}
DEV float gmax16(float v){           // reduce within 16-lane half (masks<16 stay in half)
  #pragma unroll
  for (int m = 1; m < 16; m <<= 1) v = fmaxf(v, __shfl_xor(v, m, 32));
  return v;
}
DEV float gsum16(float v){
  #pragma unroll
  for (int m = 1; m < 16; m <<= 1) v += __shfl_xor(v, m, 32);
  return v;
}

// ---------------- WMMA fragment loaders (16x16x32 f16) ----------------
// A (16x32 MxK, row-major in LDS): lane<16: halves K 0..7,16..23 ; lane>=16: 8..15,24..31
DEV v16h frag_a(const _Float16* base, int pitch, int kbase){
  const int lane = threadIdx.x & 31;
  const int m = lane & 15, hi = lane >> 4;
  const _Float16* p = base + m * pitch + kbase + hi * 8;
  v8h c0 = *(const v8h*)p;
  v8h c1 = *(const v8h*)(p + 16);
  v16h a;
  #pragma unroll
  for (int i = 0; i < 8; ++i){ a[i] = c0[i]; a[i + 8] = c1[i]; }
  return a;
}
// B (32x16 KxN) with tile stored (N,K) row-major in LDS: lane<16: K 0..7,16..23 ; lane>=16: 8..15,24..31
DEV v16h frag_b(const _Float16* base, int pitch, int kbase){
  const int lane = threadIdx.x & 31;
  const int n = lane & 15, hi = lane >> 4;
  const _Float16* p = base + n * pitch + kbase + hi * 8;
  v8h c0 = *(const v8h*)p;
  v8h c1 = *(const v8h*)(p + 16);
  v16h b;
  #pragma unroll
  for (int i = 0; i < 8; ++i){ b[i] = c0[i]; b[i + 8] = c1[i]; }
  return b;
}
DEV v8f wmma_f16(v16h a, v16h b, v8f c){
  return __builtin_amdgcn_wmma_f32_16x16x32_f16(false, a, false, b, (short)0, c, false, false);
}

// async copy: global -> LDS (ASYNCcnt-tracked), 16B per lane
DEV void async_ld_lds_b128(const _Float16* gaddr, _Float16* lds_dst){
  unsigned int la = (unsigned int)(unsigned long long)(uintptr_t)lds_dst; // low 32 bits give the LDS address
  unsigned long long ga = (unsigned long long)(uintptr_t)gaddr;
  asm volatile("global_load_async_to_lds_b128 %0, %1, off"
               :: "v"(la), "v"(ga) : "memory");
}
DEV void wait_asynccnt0(){
  asm volatile("s_wait_asynccnt 0x0" ::: "memory");
}

// ---------------- elementwise: x fp32 -> f16 ----------------
__global__ void __launch_bounds__(256) cvt_f32_f16_kernel(const float* __restrict__ in,
                                                          _Float16* __restrict__ out){
  const size_t i = ((size_t)blockIdx.x * 256 + threadIdx.x) * 4;
  f4 v = *(const f4*)(in + i);
  v4h h;
  #pragma unroll
  for (int j = 0; j < 4; ++j) h[j] = (_Float16)v[j];
  *(v4h*)(out + i) = h;
}

// ---------------- L2-normalize weight rows (1024 wide), fp32 -> f16 ----------------
__global__ void __launch_bounds__(256) norm_rows_kernel(const float* __restrict__ W,
                                                        _Float16* __restrict__ out){
  const int wave = threadIdx.x >> 5, lane = threadIdx.x & 31;
  const int row = blockIdx.x * 8 + wave;
  const float* r = W + (size_t)row * 1024;
  f4 vals[8];
  float ss = 0.f;
  #pragma unroll
  for (int c = 0; c < 8; ++c){
    vals[c] = *(const f4*)(r + c * 128 + lane * 4);
    #pragma unroll
    for (int j = 0; j < 4; ++j) ss += vals[c][j] * vals[c][j];
  }
  ss = wsum32(ss);
  const float rn = 1.f / fmaxf(sqrtf(ss), 1e-12f);
  #pragma unroll
  for (int c = 0; c < 8; ++c){
    v4h h;
    #pragma unroll
    for (int j = 0; j < 4; ++j) h[j] = (_Float16)(vals[c][j] * rn);
    *(v4h*)(out + (size_t)row * 1024 + c * 128 + lane * 4) = h;
  }
}

// ---------------- GEMM: C(MxN) = A(MxK,f16) * B(NxK,f16)^T ----------------
// 256 thr = 8 waves, tile 128x128, K-panel 64 (pitch 72)
template<bool F32OUT>
__global__ void __launch_bounds__(256) gemm_atb_kernel(const _Float16* __restrict__ A,
                                                       const _Float16* __restrict__ B,
                                                       float* __restrict__ Cf,
                                                       _Float16* __restrict__ Ch,
                                                       int N, int K){
  const int tid = threadIdx.x, wave = tid >> 5, lane = tid & 31;
  const int n16 = lane & 15, hi = lane >> 4;
  const int tm = blockIdx.y, tn = blockIdx.x;
  const int wm = wave & 1, wn = wave >> 1;          // 2x4 wave grid: 64 rows x 32 cols / wave
  __shared__ alignas(16) _Float16 As[128 * 72];
  __shared__ alignas(16) _Float16 Bs[128 * 72];
  v8f acc[4][2] = {};
  for (int kk = 0; kk < K; kk += 64){
    __syncthreads();
    #pragma unroll
    for (int i = 0; i < 4; ++i){
      int it = tid + i * 256;                        // 1024 chunks of 8 halves
      int row = it >> 3, c = it & 7;
      *(v8h*)&As[row * 72 + c * 8] = *(const v8h*)&A[(size_t)(tm * 128 + row) * K + kk + c * 8];
      *(v8h*)&Bs[row * 72 + c * 8] = *(const v8h*)&B[(size_t)(tn * 128 + row) * K + kk + c * 8];
    }
    if (kk + 64 < K){
      __builtin_prefetch(&A[(size_t)(tm * 128 + (tid >> 1)) * K + kk + 64], 0, 1);
      __builtin_prefetch(&B[(size_t)(tn * 128 + (tid >> 1)) * K + kk + 64], 0, 1);
    }
    __syncthreads();
    #pragma unroll
    for (int kb = 0; kb < 64; kb += 32){
      v16h bf0 = frag_b(Bs + (wn * 32     ) * 72, 72, kb);
      v16h bf1 = frag_b(Bs + (wn * 32 + 16) * 72, 72, kb);
      #pragma unroll
      for (int i = 0; i < 4; ++i){
        v16h af = frag_a(As + (wm * 64 + i * 16) * 72, 72, kb);
        acc[i][0] = wmma_f16(af, bf0, acc[i][0]);
        acc[i][1] = wmma_f16(af, bf1, acc[i][1]);
      }
    }
  }
  #pragma unroll
  for (int i = 0; i < 4; ++i)
    #pragma unroll
    for (int j = 0; j < 2; ++j)
      #pragma unroll
      for (int e = 0; e < 8; ++e){
        int m = tm * 128 + wm * 64 + i * 16 + e + hi * 8;
        int n = tn * 128 + wn * 32 + j * 16 + n16;
        if (F32OUT) Cf[(size_t)m * N + n] = acc[i][j][e];
        else        Ch[(size_t)m * N + n] = (_Float16)acc[i][j][e];
      }
}

// ---------------- q/k l2norm + scale + RoPE + transpose to (b,h,n,d); v transpose ----------------
// mode: 0 = v copy, 1 = q (scale*sqrt(DIM)*sqrt(DIM_HEAD)), 2 = k (scale*sqrt(DIM))
__global__ void __launch_bounds__(256) qkv_fixup_kernel(const _Float16* __restrict__ raw,
                                                        const float* __restrict__ qk_param,
                                                        _Float16* __restrict__ out,
                                                        int mode){
  const int wave = threadIdx.x >> 5, lane = threadIdx.x & 31;
  const int g = blockIdx.x * 8 + wave;               // one wave per (row, head)
  const int r = g >> 4, h = g & 15;                  // r = b*2048 + npos
  const int b = r >> 11, npos = r & 2047;
  const _Float16* src = raw + (size_t)r * 1024 + h * 64 + lane * 2;
  _Float16* dst = out + ((size_t)(b * 16 + h) * 2048 + npos) * 64 + lane * 2;
  if (mode == 0){ *(v2h*)dst = *(const v2h*)src; return; }
  float x0 = (float)src[0], x1 = (float)src[1];
  float ss = wsum32(x0 * x0 + x1 * x1);
  float rn = 1.f / fmaxf(sqrtf(ss), 1e-12f);
  const float basescale = (mode == 1) ? 256.f : 32.f;   // 32=sqrt(1024); q folds sqrt(64)=8 too
  float s0 = qk_param[h * 64 + lane * 2]     * basescale;
  float s1 = qk_param[h * 64 + lane * 2 + 1] * basescale;
  x0 *= rn * s0; x1 *= rn * s1;
  const float invf = __expf(-0.28782313662425572f * (float)lane);  // 10000^(-2p/64)
  const float ang = (float)npos * invf;
  const float cs = __cosf(ang), sn = __sinf(ang);
  v2h o; o[0] = (_Float16)(x0 * cs - x1 * sn); o[1] = (_Float16)(x1 * cs + x0 * sn);
  *(v2h*)dst = o;
}

// ---------------- flash attention: 128-query tile per block, 8 waves ----------------
__global__ void __launch_bounds__(256) attn_kernel(const _Float16* __restrict__ Q,
                                                   const _Float16* __restrict__ K,
                                                   const _Float16* __restrict__ V,
                                                   const int* __restrict__ mask,
                                                   _Float16* __restrict__ out){
  const int qt = blockIdx.x;                // query tile 0..15 (128 queries each)
  const int bh = blockIdx.y;                // b*16 + h
  const int b = bh >> 4, h = bh & 15;
  const int tid = threadIdx.x, wave = tid >> 5, lane = tid & 31;
  const int n16 = lane & 15, hi = lane >> 4;

  __shared__ alignas(16) _Float16 Qs[128 * 72];  // (query, d)
  __shared__ alignas(16) _Float16 Ks[64 * 72];   // (key, d)   == B frag layout for Q*K^T
  __shared__ alignas(16) _Float16 Vt[64 * 72];   // (d, key)   == B frag layout for P*V
  __shared__ alignas(16) _Float16 Ps[128 * 72];  // (query, key)
  __shared__ int qm_s[128];
  __shared__ int km_s[64];

  const size_t hb = (size_t)bh * (2048 * 64);

  for (int it = tid; it < 1024; it += 256){
    int row = it >> 3, c = it & 7;
    *(v8h*)&Qs[row * 72 + c * 8] = *(const v8h*)&Q[hb + (size_t)(qt * 128 + row) * 64 + c * 8];
  }
  if (tid < 128) qm_s[tid] = mask[b * 2048 + qt * 128 + tid];
  __syncthreads();

  const v16h aq0 = frag_a(Qs + wave * 16 * 72, 72, 0);
  const v16h aq1 = frag_a(Qs + wave * 16 * 72, 72, 32);
  int qm[8];
  #pragma unroll
  for (int e = 0; e < 8; ++e) qm[e] = qm_s[wave * 16 + e + hi * 8];

  v8f oacc[4] = {};
  float mrun[8], lrun[8];
  #pragma unroll
  for (int e = 0; e < 8; ++e){ mrun[e] = -1e30f; lrun[e] = 0.f; }

  for (int kt = 0; kt < 32; ++kt){
    __syncthreads();
    // K tile: async global->LDS copy (ASYNCcnt path)
    #pragma unroll
    for (int i = 0; i < 2; ++i){
      int it = tid + i * 256;                        // 512 chunks of 8 halves
      int row = it >> 3, c = it & 7;
      async_ld_lds_b128(&K[hb + (size_t)(kt * 64 + row) * 64 + c * 8], &Ks[row * 72 + c * 8]);
    }
    // V tile: through VGPRs, transposed into (d, key)
    #pragma unroll
    for (int i = 0; i < 2; ++i){
      int it = tid + i * 256;
      int row = it >> 3, c = it & 7;
      v8h vv = *(const v8h*)&V[hb + (size_t)(kt * 64 + row) * 64 + c * 8];
      #pragma unroll
      for (int u = 0; u < 8; ++u) Vt[(c * 8 + u) * 72 + row] = vv[u];
    }
    if (tid < 64) km_s[tid] = mask[b * 2048 + kt * 64 + tid];
    if (kt + 1 < 32){
      __builtin_prefetch(&K[hb + (size_t)((kt + 1) * 64) * 64], 0, 1);
      __builtin_prefetch(&V[hb + (size_t)((kt + 1) * 64) * 64], 0, 1);
    }
    wait_asynccnt0();
    __syncthreads();

    // S = Q * K^T  (score scale sqrt(64) pre-folded into Q)
    v8f sacc[4] = {};
    #pragma unroll
    for (int t = 0; t < 4; ++t){
      sacc[t] = wmma_f16(aq0, frag_b(Ks + t * 16 * 72, 72, 0),  sacc[t]);
      sacc[t] = wmma_f16(aq1, frag_b(Ks + t * 16 * 72, 72, 32), sacc[t]);
    }

    // mask + online softmax in C layout: row = e + hi*8, col = t*16 + n16
    float rm[8];
    #pragma unroll
    for (int e = 0; e < 8; ++e) rm[e] = -1e30f;
    int kmv[4];
    #pragma unroll
    for (int t = 0; t < 4; ++t){
      kmv[t] = km_s[t * 16 + n16];
      #pragma unroll
      for (int e = 0; e < 8; ++e){
        float sv = (qm[e] == kmv[t]) ? sacc[t][e] : -1e30f;
        sacc[t][e] = sv;
        rm[e] = fmaxf(rm[e], sv);
      }
    }
    float mnew[8], alpha[8];
    #pragma unroll
    for (int e = 0; e < 8; ++e){
      mnew[e] = fmaxf(mrun[e], gmax16(rm[e]));
      alpha[e] = __expf(mrun[e] - mnew[e]);
      mrun[e] = mnew[e];
    }
    float rs[8] = {0.f,0.f,0.f,0.f,0.f,0.f,0.f,0.f};
    #pragma unroll
    for (int t = 0; t < 4; ++t){
      #pragma unroll
      for (int e = 0; e < 8; ++e){
        float p = (qm[e] == kmv[t]) ? __expf(sacc[t][e] - mnew[e]) : 0.f;
        rs[e] += p;
        Ps[(wave * 16 + e + hi * 8) * 72 + t * 16 + n16] = (_Float16)p;
      }
    }
    #pragma unroll
    for (int e = 0; e < 8; ++e) lrun[e] = lrun[e] * alpha[e] + gsum16(rs[e]);
    #pragma unroll
    for (int t = 0; t < 4; ++t)
      #pragma unroll
      for (int e = 0; e < 8; ++e) oacc[t][e] *= alpha[e];

    __syncthreads();   // order Ps stores before A-frag reloads
    const v16h ap0 = frag_a(Ps + wave * 16 * 72, 72, 0);
    const v16h ap1 = frag_a(Ps + wave * 16 * 72, 72, 32);
    #pragma unroll
    for (int t = 0; t < 4; ++t){
      oacc[t] = wmma_f16(ap0, frag_b(Vt + t * 16 * 72, 72, 0),  oacc[t]);
      oacc[t] = wmma_f16(ap1, frag_b(Vt + t * 16 * 72, 72, 32), oacc[t]);
    }
  }

  float inv[8];
  #pragma unroll
  for (int e = 0; e < 8; ++e) inv[e] = 1.f / lrun[e];
  #pragma unroll
  for (int t = 0; t < 4; ++t)
    #pragma unroll
    for (int e = 0; e < 8; ++e){
      int q = qt * 128 + wave * 16 + e + hi * 8;
      int col = h * 64 + t * 16 + n16;
      out[(size_t)(b * 2048 + q) * 1024 + col] = (_Float16)(oacc[t][e] * inv[e]);
    }
}

// ---------------- launcher ----------------
extern "C" void kernel_launch(void* const* d_in, const int* in_sizes, int n_in,
                              void* d_out, int out_size, void* d_ws, size_t ws_size,
                              hipStream_t stream){
  const float* x    = (const float*)d_in[0];
  const int*   mask = (const int*)d_in[1];
  const float* Wq   = (const float*)d_in[2];
  const float* Wk   = (const float*)d_in[3];
  const float* Wv   = (const float*)d_in[4];
  const float* Wo   = (const float*)d_in[5];
  const float* qk   = (const float*)d_in[6];
  float* out = (float*)d_out;
  char* ws = (char*)d_ws;
  const size_t MB = 1024ull * 1024ull;

  _Float16* Wq16 = (_Float16*)(ws + 0 * MB);
  _Float16* Wk16 = (_Float16*)(ws + 2 * MB);
  _Float16* Wv16 = (_Float16*)(ws + 4 * MB);
  _Float16* Wo16 = (_Float16*)(ws + 6 * MB);
  _Float16* X16  = (_Float16*)(ws + 8 * MB);
  _Float16* Qraw = (_Float16*)(ws + 16 * MB);
  _Float16* Kraw = (_Float16*)(ws + 24 * MB);
  _Float16* Vraw = (_Float16*)(ws + 32 * MB);
  _Float16* Qa   = (_Float16*)(ws + 40 * MB);
  _Float16* Ka   = (_Float16*)(ws + 48 * MB);
  _Float16* Va   = (_Float16*)(ws + 56 * MB);
  _Float16* AO16 = Qraw;                    // Qraw dead after fixup -> reuse

  cvt_f32_f16_kernel<<<4096, 256, 0, stream>>>(x, X16);
  norm_rows_kernel<<<128, 256, 0, stream>>>(Wq, Wq16);
  norm_rows_kernel<<<128, 256, 0, stream>>>(Wk, Wk16);
  norm_rows_kernel<<<128, 256, 0, stream>>>(Wv, Wv16);
  norm_rows_kernel<<<128, 256, 0, stream>>>(Wo, Wo16);

  dim3 gg(1024 / 128, 4096 / 128);
  gemm_atb_kernel<false><<<gg, 256, 0, stream>>>(X16, Wq16, nullptr, Qraw, 1024, 1024);
  gemm_atb_kernel<false><<<gg, 256, 0, stream>>>(X16, Wk16, nullptr, Kraw, 1024, 1024);
  gemm_atb_kernel<false><<<gg, 256, 0, stream>>>(X16, Wv16, nullptr, Vraw, 1024, 1024);

  qkv_fixup_kernel<<<8192, 256, 0, stream>>>(Qraw, qk, Qa, 1);
  qkv_fixup_kernel<<<8192, 256, 0, stream>>>(Kraw, qk, Ka, 2);
  qkv_fixup_kernel<<<8192, 256, 0, stream>>>(Vraw, qk, Va, 0);

  attn_kernel<<<dim3(16, 32), 256, 0, stream>>>(Qa, Ka, Va, mask, AO16);

  gemm_atb_kernel<true><<<gg, 256, 0, stream>>>(AO16, Wo16, out, nullptr, 1024, 1024);

  (void)in_sizes; (void)n_in; (void)out_size; (void)ws_size;
}
